// HorizontallySplitAttention_592705487370
// MI455X (gfx1250) — matrix operand order repaired and checked by reading, not compiled, not run on hardware
//
#include <hip/hip_runtime.h>
#include <hip/hip_bf16.h>
#include <stdint.h>

// ---------------- problem constants ----------------
#define BSZ   2
#define SEQ   2048
#define DIMSZ 4096
#define NHEAD 32
#define NKVH  8
#define HDIM  128
#define NREP  (NHEAD / NKVH)
#define MROWS (BSZ * SEQ)          // 4096
#define QCOLS (NHEAD * HDIM)       // 4096
#define KVCOLS (NKVH * HDIM)       // 1024

typedef __bf16 bf16;
typedef __attribute__((ext_vector_type(16))) __bf16 v16bf;
typedef __attribute__((ext_vector_type(8)))  float  v8f;
typedef __attribute__((ext_vector_type(4)))  unsigned int u32x4;
typedef __attribute__((ext_vector_type(8)))  int i32x8;
typedef __attribute__((ext_vector_type(4)))  int i32x4;

union Frag { uint32_t u[8]; v16bf v; };

__device__ inline v8f wmma_bf16(v16bf a, v16bf b, v8f c) {
  // D = A(16x32 bf16) x B(32x16 bf16) + C(16x16 f32)
  return __builtin_amdgcn_wmma_f32_16x16x32_bf16(
      /*neg_a=*/false, a, /*neg_b=*/false, b,
      /*c_mod=*/(short)0, c, /*reuse_a=*/false, /*reuse_b=*/false);
}

// K index inside a 16x32 A fragment for (vgpr i, lane-half h): ISA 7.12.2
__device__ inline int kA(int i, int h) { return ((i < 4) ? 0 : 16) + h * 8 + (i & 3) * 2; }
// K index inside a 32x16 B fragment (same K order as A fragment)
__device__ inline int kB(int i, int h) { return ((i < 4) ? 0 : 16) + h * 8 + (i & 3) * 2; }

__device__ inline uint32_t lds_off(const void* p) { return (uint32_t)(uintptr_t)p; }

// ---- Tensor Data Mover: 2D tile load, bf16 elements (ISA 08_async_tensor D#) ----
// tile_w elems per row (contiguous), tile_h rows, row_stride elems between rows.
// Destination LDS layout: contiguous [tile_h][tile_w].
__device__ inline void tdm_load_2d_bf16(const void* gaddr, uint32_t ldsaddr,
                                        uint32_t tile_w, uint32_t tile_h,
                                        uint32_t row_stride) {
  uint64_t ga = (uint64_t)(uintptr_t)gaddr;
  u32x4 g0;
  g0[0] = 1u;                                              // count=1 (valid D#)
  g0[1] = ldsaddr;                                         // lds_addr (bytes)
  g0[2] = (uint32_t)(ga & 0xFFFFFFFFu);                    // global_addr[31:0]
  g0[3] = (uint32_t)((ga >> 32) & 0x1FFFFFFu) | (2u << 30);// global_addr[56:32] | type=2
  i32x8 g1;
  g1[0] = (int)(1u << 16);                                 // data_size=1 (2 bytes), mask=0
  g1[1] = (int)((tile_w & 0xFFFFu) << 16);                 // tensor_dim0[15:0] @ bits63:48
  g1[2] = (int)((tile_w >> 16) | ((tile_h & 0xFFFFu) << 16)); // dim0 hi | tensor_dim1 lo
  g1[3] = (int)((tile_h >> 16) | (tile_w << 16));          // dim1 hi | tile_dim0
  g1[4] = (int)(tile_h & 0xFFFFu);                         // tile_dim1 (tile_dim2=0)
  g1[5] = (int)row_stride;                                 // tensor_dim0_stride[31:0]
  g1[6] = 0;                                               // stride hi | dim1_stride lo (2D)
  g1[7] = 0;
  i32x4 z4 = {0, 0, 0, 0};
  i32x8 z8 = {0, 0, 0, 0, 0, 0, 0, 0};
  __builtin_amdgcn_tensor_load_to_lds(g0, g1, z4, z4, z8, 0);
}

// ---------------- fp32 -> bf16 conversion ----------------
__global__ __launch_bounds__(256) void cvt_f32_bf16(const float* __restrict__ src,
                                                    bf16* __restrict__ dst, size_t n) {
  size_t i = ((size_t)blockIdx.x * blockDim.x + threadIdx.x) * 4;
  if (i >= n) return;
  float4 v = *(const float4*)(src + i);
  dst[i + 0] = (bf16)v.x;
  dst[i + 1] = (bf16)v.y;
  dst[i + 2] = (bf16)v.z;
  dst[i + 3] = (bf16)v.w;
}

// ---------------- in-place RoPE on bf16 (B,S,nh,HD) ----------------
__global__ __launch_bounds__(256) void rope_kernel(bf16* __restrict__ t,
                                                   const float* __restrict__ fcos,
                                                   const float* __restrict__ fsin,
                                                   int nh) {
  size_t idx = (size_t)blockIdx.x * blockDim.x + threadIdx.x;
  size_t total = (size_t)BSZ * SEQ * nh * (HDIM / 2);
  if (idx >= total) return;
  int j = (int)(idx % (HDIM / 2));
  int hh = (int)((idx / (HDIM / 2)) % nh);
  int s = (int)((idx / ((size_t)(HDIM / 2) * nh)) % SEQ);
  int b = (int)(idx / ((size_t)(HDIM / 2) * nh * SEQ));
  float c = fcos[s * (HDIM / 2) + j];
  float sn = fsin[s * (HDIM / 2) + j];
  bf16* p = t + (((size_t)(b * SEQ + s)) * nh + hh) * HDIM + 2 * j;
  float tr = (float)p[0], ti = (float)p[1];
  p[0] = (bf16)(tr * c - ti * sn);
  p[1] = (bf16)(tr * sn + ti * c);
}

// ---------------- bf16 GEMM: C(MxN) = A(MxK) * W(KxN) ----------------
// 128x128 block tile, K-depth 64 per stage. A tile loaded by TDM (wave 0),
// W tile cooperatively transposed into LDS. 8 waves, wave = 32(M) x 64(N),
// 16 WMMA per stage.
template <bool F32OUT>
__global__ __launch_bounds__(256) void gemm_bf16_kernel(const bf16* __restrict__ A,
                                                        const bf16* __restrict__ W,
                                                        void* __restrict__ Cv,
                                                        int M, int N, int K) {
  __shared__ bf16 ldsA[128 * 64];   // [m][k]
  __shared__ bf16 ldsB[128 * 64];   // [n][k]  (transposed so K-pairs are contiguous)

  const int tid = threadIdx.x;
  const int lane = tid & 31;
  const int wave = tid >> 5;
  const int waveM = wave & 3;   // 4 slices of 32 rows
  const int waveN = wave >> 2;  // 2 slices of 64 cols
  const int half = lane >> 4;
  const int l16 = lane & 15;
  const int m0 = blockIdx.y * 128;
  const int n0 = blockIdx.x * 128;

  v8f acc[2][4];
#pragma unroll
  for (int mi = 0; mi < 2; ++mi)
#pragma unroll
    for (int ni = 0; ni < 4; ++ni) acc[mi][ni] = (v8f){0, 0, 0, 0, 0, 0, 0, 0};

  for (int k0 = 0; k0 < K; k0 += 64) {
    // A tile 128x64 via Tensor Data Mover (single wave issues the DMA)
    if (wave == 0) {
      tdm_load_2d_bf16(A + (size_t)m0 * K + k0, lds_off(ldsA),
                       /*tile_w=*/64, /*tile_h=*/128, /*row_stride=*/(uint32_t)K);
    }
    // W tile 64x128, cooperatively store transposed [n][k] (overlaps with TDM)
#pragma unroll
    for (int i = 0; i < 4; ++i) {
      int idx = tid + i * 256;       // 1024 uint4
      int row = idx >> 4;            // k row (0..63)
      int c8 = (idx & 15) << 3;      // n start
      uint4 d = *(const uint4*)(W + (size_t)(k0 + row) * N + n0 + c8);
      const bf16* e = (const bf16*)&d;
#pragma unroll
      for (int j = 0; j < 8; ++j) ldsB[(c8 + j) * 64 + row] = e[j];
    }
    if (wave == 0) __builtin_amdgcn_s_wait_tensorcnt(0);
    __syncthreads();

#pragma unroll
    for (int kk = 0; kk < 2; ++kk) {
      Frag af[2], bfr[4];
#pragma unroll
      for (int mi = 0; mi < 2; ++mi)
#pragma unroll
        for (int i = 0; i < 8; ++i)
          af[mi].u[i] = *(const uint32_t*)&ldsA[(waveM * 32 + mi * 16 + l16) * 64 +
                                                kk * 32 + kA(i, half)];
#pragma unroll
      for (int ni = 0; ni < 4; ++ni)
#pragma unroll
        for (int i = 0; i < 8; ++i)
          bfr[ni].u[i] = *(const uint32_t*)&ldsB[(waveN * 64 + ni * 16 + l16) * 64 +
                                                 kk * 32 + kB(i, half)];
#pragma unroll
      for (int mi = 0; mi < 2; ++mi)
#pragma unroll
        for (int ni = 0; ni < 4; ++ni)
          acc[mi][ni] = wmma_bf16(af[mi].v, bfr[ni].v, acc[mi][ni]);
    }
    __syncthreads();
  }

  // epilogue: C-layout -> global
#pragma unroll
  for (int mi = 0; mi < 2; ++mi)
#pragma unroll
    for (int ni = 0; ni < 4; ++ni)
#pragma unroll
      for (int r = 0; r < 8; ++r) {
        int row = m0 + waveM * 32 + mi * 16 + half * 8 + r;
        int col = n0 + waveN * 64 + ni * 16 + l16;
        float v = acc[mi][ni][r];
        if (F32OUT)
          ((float*)Cv)[(size_t)row * N + col] = v;
        else
          ((bf16*)Cv)[(size_t)row * N + col] = (bf16)v;
      }
}

// ---------------- flash attention ----------------
// grid: (B*NH, S/128). block: 256 thr = 8 waves, wave owns 16 q-rows.
// K tile via TDM; V tile cooperatively transposed; online softmax; all WMMA bf16.
__global__ __launch_bounds__(256) void flash_attn_kernel(const bf16* __restrict__ Q,
                                                         const bf16* __restrict__ Kc,
                                                         const bf16* __restrict__ Vc,
                                                         bf16* __restrict__ Oa) {
  __shared__ bf16 ldsK[64 * 128];     // [key][hd]
  __shared__ bf16 ldsVt[128 * 64];    // [hd][key]
  __shared__ bf16 ldsP[8][16 * 64];   // per-wave P (16 q-rows x 64 keys)

  const int tid = threadIdx.x;
  const int lane = tid & 31;
  const int wave = tid >> 5;
  const int half = lane >> 4;
  const int l16 = lane & 15;
  const int b = blockIdx.x / NHEAD;
  const int h = blockIdx.x % NHEAD;
  const int hk = h / NREP;
  const int q0 = blockIdx.y * 128;
  const int qw = q0 + wave * 16;

  // preload Q fragments (A-layout), 4 chunks of hd
  Frag qf[4];
  {
    const bf16* qptr = Q + (((size_t)(b * SEQ + qw + l16)) * NHEAD + h) * HDIM;
#pragma unroll
    for (int c = 0; c < 4; ++c)
#pragma unroll
      for (int i = 0; i < 8; ++i)
        qf[c].u[i] = *(const uint32_t*)(qptr + c * 32 + kA(i, half));
  }

  v8f o_acc[8];
#pragma unroll
  for (int os = 0; os < 8; ++os) o_acc[os] = (v8f){0, 0, 0, 0, 0, 0, 0, 0};
  float m_r[8], l_r[8];
#pragma unroll
  for (int r = 0; r < 8; ++r) { m_r[r] = -1e30f; l_r[r] = 0.f; }

  const float scale = 0.08838834764831845f;  // 1/sqrt(128)
  const int nkb = (q0 + 128) / 64;

  for (int kb = 0; kb < nkb; ++kb) {
    const int kbase = kb * 64;
    // K tile [64][128] via TDM (wave 0); row stride NKVH*HDIM
    if (wave == 0) {
      tdm_load_2d_bf16(Kc + (((size_t)(b * SEQ + kbase)) * NKVH + hk) * HDIM,
                       lds_off(ldsK), /*tile_w=*/HDIM, /*tile_h=*/64,
                       /*row_stride=*/NKVH * HDIM);
    }
    // V tile transposed [hd][key] cooperatively (overlaps with TDM)
#pragma unroll
    for (int i = 0; i < 4; ++i) {
      int idx = tid + i * 256;
      int row = idx >> 4;
      int c8 = (idx & 15) << 3;
      uint4 d = *(const uint4*)(Vc + (((size_t)(b * SEQ + kbase + row)) * NKVH + hk) * HDIM + c8);
      const bf16* e = (const bf16*)&d;
#pragma unroll
      for (int j = 0; j < 8; ++j) ldsVt[(c8 + j) * 64 + row] = e[j];
    }
    if (wave == 0) __builtin_amdgcn_s_wait_tensorcnt(0);
    __syncthreads();

    // S = Q @ K^T  (4 key-subtiles x 4 hd-chunks = 16 WMMA)
    v8f s_acc[4];
#pragma unroll
    for (int ks = 0; ks < 4; ++ks) s_acc[ks] = (v8f){0, 0, 0, 0, 0, 0, 0, 0};
#pragma unroll
    for (int ks = 0; ks < 4; ++ks) {
#pragma unroll
      for (int c = 0; c < 4; ++c) {
        Frag bfr;
#pragma unroll
        for (int i = 0; i < 8; ++i)
          bfr.u[i] = *(const uint32_t*)&ldsK[(ks * 16 + l16) * 128 + c * 32 + kB(i, half)];
        s_acc[ks] = wmma_bf16(qf[c].v, bfr.v, s_acc[ks]);
      }
    }

    // causal mask + online softmax (row = half*8 + r, col lanes within half)
#pragma unroll
    for (int r = 0; r < 8; ++r) {
      const int qrow = qw + half * 8 + r;
      float sv[4];
      float mx = -1e30f;
#pragma unroll
      for (int ks = 0; ks < 4; ++ks) {
        int key = kbase + ks * 16 + l16;
        float s = s_acc[ks][r] * scale;
        if (key > qrow) s = -1e9f;
        sv[ks] = s;
        mx = fmaxf(mx, s);
      }
      for (int off = 8; off; off >>= 1) mx = fmaxf(mx, __shfl_xor(mx, off, 16));
      const float mnew = fmaxf(m_r[r], mx);
      const float corr = __expf(m_r[r] - mnew);
      float rs = 0.f;
#pragma unroll
      for (int ks = 0; ks < 4; ++ks) {
        float p = __expf(sv[ks] - mnew);
        rs += p;
        s_acc[ks][r] = p;
      }
      for (int off = 8; off; off >>= 1) rs += __shfl_xor(rs, off, 16);
      l_r[r] = l_r[r] * corr + rs;
      m_r[r] = mnew;
#pragma unroll
      for (int os = 0; os < 8; ++os) o_acc[os][r] *= corr;
    }

    // P (C-layout f32) -> LDS bf16 (row-major 16x64)
#pragma unroll
    for (int ks = 0; ks < 4; ++ks)
#pragma unroll
      for (int r = 0; r < 8; ++r)
        ldsP[wave][(half * 8 + r) * 64 + ks * 16 + l16] = (bf16)s_acc[ks][r];
    __syncthreads();

    // O += P @ V  (8 hd-subtiles x 2 key-chunks = 16 WMMA)
    Frag pf[2];
#pragma unroll
    for (int c = 0; c < 2; ++c)
#pragma unroll
      for (int i = 0; i < 8; ++i)
        pf[c].u[i] = *(const uint32_t*)&ldsP[wave][l16 * 64 + c * 32 + kA(i, half)];
#pragma unroll
    for (int os = 0; os < 8; ++os) {
#pragma unroll
      for (int c = 0; c < 2; ++c) {
        Frag bfr;
#pragma unroll
        for (int i = 0; i < 8; ++i)
          bfr.u[i] = *(const uint32_t*)&ldsVt[(os * 16 + l16) * 64 + c * 32 + kB(i, half)];
        o_acc[os] = wmma_bf16(pf[c].v, bfr.v, o_acc[os]);
      }
    }
    __syncthreads();
  }

  // epilogue: O/l -> attn-out bf16 (B,S,NH*HD)
#pragma unroll
  for (int os = 0; os < 8; ++os)
#pragma unroll
    for (int r = 0; r < 8; ++r) {
      int qrow = qw + half * 8 + r;
      float v = o_acc[os][r] / l_r[r];
      Oa[((size_t)(b * SEQ + qrow)) * QCOLS + h * HDIM + os * 16 + l16] = (bf16)v;
    }
}

// ---------------- launch ----------------
extern "C" void kernel_launch(void* const* d_in, const int* in_sizes, int n_in,
                              void* d_out, int out_size, void* d_ws, size_t ws_size,
                              hipStream_t stream) {
  (void)in_sizes; (void)n_in; (void)out_size; (void)ws_size;
  const float* x  = (const float*)d_in[0];
  const float* wq = (const float*)d_in[1];
  const float* wk = (const float*)d_in[2];
  const float* wv = (const float*)d_in[3];
  const float* wo = (const float*)d_in[4];
  const float* fc = (const float*)d_in[5];
  const float* fs = (const float*)d_in[6];
  float* out = (float*)d_out;

  char* ws = (char*)d_ws;
  const size_t SZ_XB  = (size_t)MROWS * DIMSZ * 2;   // 32MB
  const size_t SZ_WQ  = (size_t)DIMSZ * QCOLS * 2;   // 32MB
  const size_t SZ_WKV = (size_t)DIMSZ * KVCOLS * 2;  // 8MB
  const size_t SZ_Q   = (size_t)MROWS * QCOLS * 2;   // 32MB
  const size_t SZ_KV  = (size_t)MROWS * KVCOLS * 2;  // 8MB

  bf16* xb  = (bf16*)(ws);                       size_t off = SZ_XB;
  bf16* wqb = (bf16*)(ws + off);                 off += SZ_WQ;
  bf16* wkb = (bf16*)(ws + off);                 off += SZ_WKV;
  bf16* wvb = (bf16*)(ws + off);                 off += SZ_WKV;
  bf16* wob = (bf16*)(ws + off);                 off += SZ_WQ;
  bf16* qb  = (bf16*)(ws + off);                 off += SZ_Q;
  bf16* kb  = (bf16*)(ws + off);                 off += SZ_KV;
  bf16* vb  = (bf16*)(ws + off);                 off += SZ_KV;
  bf16* ab  = (bf16*)(ws + off);                 off += SZ_Q;

  auto cvt = [&](const float* s, bf16* d, size_t n) {
    cvt_f32_bf16<<<(unsigned)((n / 4 + 255) / 256), 256, 0, stream>>>(s, d, n);
  };
  cvt(x,  xb,  (size_t)MROWS * DIMSZ);
  cvt(wq, wqb, (size_t)DIMSZ * QCOLS);
  cvt(wk, wkb, (size_t)DIMSZ * KVCOLS);
  cvt(wv, wvb, (size_t)DIMSZ * KVCOLS);
  cvt(wo, wob, (size_t)DIMSZ * QCOLS);

  // projections (bf16 out)
  gemm_bf16_kernel<false><<<dim3(QCOLS / 128, MROWS / 128), 256, 0, stream>>>(
      xb, wqb, qb, MROWS, QCOLS, DIMSZ);
  gemm_bf16_kernel<false><<<dim3(KVCOLS / 128, MROWS / 128), 256, 0, stream>>>(
      xb, wkb, kb, MROWS, KVCOLS, DIMSZ);
  gemm_bf16_kernel<false><<<dim3(KVCOLS / 128, MROWS / 128), 256, 0, stream>>>(
      xb, wvb, vb, MROWS, KVCOLS, DIMSZ);

  // RoPE in place on Q and K
  {
    size_t nq = (size_t)BSZ * SEQ * NHEAD * (HDIM / 2);
    rope_kernel<<<(unsigned)((nq + 255) / 256), 256, 0, stream>>>(qb, fc, fs, NHEAD);
    size_t nk = (size_t)BSZ * SEQ * NKVH * (HDIM / 2);
    rope_kernel<<<(unsigned)((nk + 255) / 256), 256, 0, stream>>>(kb, fc, fs, NKVH);
  }

  // flash attention
  flash_attn_kernel<<<dim3(BSZ * NHEAD, SEQ / 128), 256, 0, stream>>>(qb, kb, vb, ab);

  // output projection (fp32 out)
  gemm_bf16_kernel<true><<<dim3(DIMSZ / 128, MROWS / 128), 256, 0, stream>>>(
      ab, wob, out, MROWS, DIMSZ, QCOLS);
}
